// BilinearInteraction_15745350107446
// MI455X (gfx1250) — matrix operand order repaired and checked by reading, not compiled, not run on hardware
//
#include <hip/hip_runtime.h>
#include <cstdint>
#include <cstddef>

// Problem geometry (fixed by the reference).
#define BATCH      16384
#define NUM_FIELDS 40
#define EMBED_DIM  32
#define KDIM       (NUM_FIELDS * EMBED_DIM)   // 1280 (reduction dim, g*32+e)
#define NDIM       (NUM_FIELDS * EMBED_DIM)   // 1280 (output dim,   f*32+d)
#define KSTEPS     (KDIM / 32)                // 40

// CDNA5 WMMA operand vectors (wave32).
typedef __attribute__((ext_vector_type(16))) __bf16        v16bf;
typedef __attribute__((ext_vector_type(8)))  float         v8f;
typedef __attribute__((ext_vector_type(4)))  unsigned int  u32x4;
typedef __attribute__((ext_vector_type(8)))  unsigned int  u32x8;

// 16 bf16 fragment viewed as two 16-byte load quanta / raw shorts.
union Frag16 {
    v16bf          bf;
    uint4          q[2];
    unsigned short u[16];
};

__device__ __forceinline__ unsigned short f32_to_bf16_rn(float x) {
    unsigned int u = __float_as_uint(x);
    unsigned int r = 0x7FFFu + ((u >> 16) & 1u);
    return (unsigned short)((u + r) >> 16);
}
__device__ __forceinline__ float bf16_to_f32(unsigned short h) {
    return __uint_as_float(((unsigned int)h) << 16);
}

// ---------------------------------------------------------------------------
// TDM: DMA a 2D bf16 tile [rows=256, k=32] (row stride KDIM) into LDS.
// D# built per CDNA5 ISA §8 (group0: 4 SGPRs, group1: 8 SGPRs; 2D tensor so
// groups 2/3 omitted). Issued once per wave; tracked with TENSORcnt.
// ---------------------------------------------------------------------------
__device__ __forceinline__ void tdm_load_panel(const unsigned short* gsrc,
                                               unsigned int lds_byte_addr) {
    unsigned long long ga = (unsigned long long)gsrc;
    u32x4 g0;
    g0[0] = 1u;                                       // count=1, user descriptor
    g0[1] = lds_byte_addr;                            // lds_addr (bytes)
    g0[2] = (unsigned int)ga;                         // global_addr[31:0]
    g0[3] = ((unsigned int)(ga >> 32) & 0x01FFFFFFu)  // global_addr[56:32]
            | (2u << 30);                             // type=2 ("image")
    u32x8 g1;
    g1[0] = 0x00010000u;                              // wg_mask=0, data_size=2B
    g1[1] = ((unsigned)KDIM & 0xFFFFu) << 16;         // tensor_dim0[15:0]
    g1[2] = ((unsigned)KDIM >> 16)                    // tensor_dim0[31:16]
            | (((unsigned)NDIM & 0xFFFFu) << 16);     // tensor_dim1[15:0]
    g1[3] = ((unsigned)NDIM >> 16) | (32u << 16);     // dim1 hi | tile_dim0=32
    g1[4] = 256u;                                     // tile_dim1=256, tile_dim2=0
    g1[5] = (unsigned)KDIM;                           // tensor_dim0_stride[31:0]
    g1[6] = 0u;                                       // stride hi | dim1_stride lo
    g1[7] = 0u;                                       // dim1_stride hi
    asm volatile("tensor_load_to_lds %0, %1" :: "s"(g0), "s"(g1) : "memory");
}

// ---------------------------------------------------------------------------
// Kernel 1: masked, transposed, hi/lo-split W -> workspace.
//   Wt[n][k] = (f==g) ? 0 : W[f][g][d][e],  n = f*32+d, k = g*32+e
// ---------------------------------------------------------------------------
__global__ void prep_w_kernel(const float* __restrict__ W,
                              unsigned short* __restrict__ WtHi,
                              unsigned short* __restrict__ WtLo) {
    int idx = blockIdx.x * blockDim.x + threadIdx.x;
    if (idx >= KDIM * NDIM) return;
    int n = idx / KDIM;
    int k = idx - n * KDIM;
    int f = n >> 5, d = n & 31;
    int g = k >> 5, e = k & 31;
    float v = 0.0f;
    if (f != g) {
        size_t off = ((((size_t)f * NUM_FIELDS + g) << 5) + (size_t)d) * EMBED_DIM + e;
        v = W[off];
    }
    unsigned short hi = f32_to_bf16_rn(v);
    unsigned short lo = f32_to_bf16_rn(v - bf16_to_f32(hi));
    WtHi[idx] = hi;
    WtLo[idx] = lo;
}

// ---------------------------------------------------------------------------
// Kernel 2: pre-split embeddings into bf16 hi/lo planes (memory-bound).
// ---------------------------------------------------------------------------
__global__ void prep_emb_kernel(const float* __restrict__ emb,
                                unsigned short* __restrict__ AHi,
                                unsigned short* __restrict__ ALo) {
    size_t idx = (size_t)blockIdx.x * blockDim.x + threadIdx.x;
    if (idx >= (size_t)BATCH * KDIM) return;
    float v = emb[idx];
    unsigned short hi = f32_to_bf16_rn(v);
    unsigned short lo = f32_to_bf16_rn(v - bf16_to_f32(hi));
    AHi[idx] = hi;
    ALo[idx] = lo;
}

// ---------------------------------------------------------------------------
// Kernel 3 (main): S = A * Wt^T via v_wmma_f32_16x16x32_bf16 (2-term bf16
// split: hi*hi + hi*lo + lo*hi, f32 accumulate), out = emb .* S.
//
// Block: 256 threads = 8 waves arranged 2 (M) x 4 (N); wave tile 32x64.
// Workgroup tile: 64 (M) x 256 (N). Grid: (1280/256, 16384/64) = (5, 256).
// B panel (256 cols x 32 k, hi+lo = 32 KB) staged to LDS per K-step by the
// Tensor Data Mover, double-buffered (64 KB LDS). Wave 0 issues/waits TDM;
// workgroup barrier publishes the panel. A streams via global_load_b128.
// ---------------------------------------------------------------------------
__global__ __launch_bounds__(256)
void bilinear_wmma_tdm_kernel(const float* __restrict__ emb,
                              const unsigned short* __restrict__ AHi,
                              const unsigned short* __restrict__ ALo,
                              const unsigned short* __restrict__ WtHi,
                              const unsigned short* __restrict__ WtLo,
                              float* __restrict__ out) {
    // [buffer][plane hi/lo][row 0..255][k-chunk 0..3]  (uint4 = 8 bf16)
    __shared__ uint4 smem[2][2][256 * 4];

    const int lane   = threadIdx.x & 31;
    const int wave   = threadIdx.x >> 5;   // 0..7
    const int wm     = wave >> 2;          // 0..1 (M)
    const int wn     = wave & 3;           // 0..3 (N)
    const int lane15 = lane & 15;
    const int laneHi = lane >> 4;          // 0 or 1

    const int m_wave = blockIdx.y * 64 + wm * 32;
    const int n_blk  = blockIdx.x * 256;

    const v8f vzero = {0.f, 0.f, 0.f, 0.f, 0.f, 0.f, 0.f, 0.f};
    v8f acc[2][4];
#pragma unroll
    for (int mi = 0; mi < 2; ++mi)
#pragma unroll
        for (int ni = 0; ni < 4; ++ni) acc[mi][ni] = vzero;

    // Per-lane K chunk bases (ISA §7.12.2, wave32):
    //   A 16-bit 16x32: lane<16 -> K {0..7} U {16..23}; lane>=16 -> +8
    //   B 16-bit 32x16: lane<16 -> K {0..7} U {16..23}; lane>=16 -> +8
    const int a_kc    = laneHi * 8;
    const int b_chunk = laneHi;            // uint4 index of first 8-element chunk

    const unsigned short* aRowH[2];
#pragma unroll
    for (int mi = 0; mi < 2; ++mi)
        aRowH[mi] = AHi + (size_t)(m_wave + mi * 16 + lane15) * KDIM + a_kc;
    const size_t aLoOff = (size_t)ALo - (size_t)AHi;

    const unsigned short* bSrcH = WtHi + (size_t)n_blk * KDIM;
    const unsigned short* bSrcL = WtLo + (size_t)n_blk * KDIM;

    // Prologue: stage panel for k=0 into buffer 0.
    if (wave == 0) {
        tdm_load_panel(bSrcH, (unsigned int)(size_t)&smem[0][0][0]);
        tdm_load_panel(bSrcL, (unsigned int)(size_t)&smem[0][1][0]);
    }

    for (int it = 0; it < KSTEPS; ++it) {
        const int k   = it * 32;
        const int buf = it & 1;

        if (wave == 0) __builtin_amdgcn_s_wait_tensorcnt(0);  // panel 'buf' ready
        __syncthreads();                                      // publish to all waves
        if (wave == 0 && it + 1 < KSTEPS) {                   // prefetch next panel
            tdm_load_panel(bSrcH + (k + 32), (unsigned int)(size_t)&smem[buf ^ 1][0][0]);
            tdm_load_panel(bSrcL + (k + 32), (unsigned int)(size_t)&smem[buf ^ 1][1][0]);
        }

        // ---- A fragments (pre-split bf16, global): 2 x b128 per fragment
        Frag16 aHi[2], aLo[2], bHi[4], bLo[4];
#pragma unroll
        for (int mi = 0; mi < 2; ++mi) {
            const unsigned short* ph = aRowH[mi] + k;
            const unsigned short* pl =
                (const unsigned short*)((const char*)ph + aLoOff);
            aHi[mi].q[0] = *(const uint4*)(ph);
            aHi[mi].q[1] = *(const uint4*)(ph + 16);
            aLo[mi].q[0] = *(const uint4*)(pl);
            aLo[mi].q[1] = *(const uint4*)(pl + 16);
        }

        // ---- B fragments from the LDS panel (ds_load_b128)
#pragma unroll
        for (int ni = 0; ni < 4; ++ni) {
            const int row = wn * 64 + ni * 16 + lane15;       // col of Wt
            bHi[ni].q[0] = smem[buf][0][row * 4 + b_chunk];
            bHi[ni].q[1] = smem[buf][0][row * 4 + b_chunk + 2];
            bLo[ni].q[0] = smem[buf][1][row * 4 + b_chunk];
            bLo[ni].q[1] = smem[buf][1][row * 4 + b_chunk + 2];
        }

        // ---- 24 WMMAs: hi*hi + hi*lo + lo*hi (lo*lo below fp32 eps)
#pragma unroll
        for (int mi = 0; mi < 2; ++mi) {
#pragma unroll
            for (int ni = 0; ni < 4; ++ni) {
                acc[mi][ni] = __builtin_amdgcn_wmma_f32_16x16x32_bf16(
                    false, aHi[mi].bf, false, bHi[ni].bf,
                    (short)0, acc[mi][ni], false, false);
                acc[mi][ni] = __builtin_amdgcn_wmma_f32_16x16x32_bf16(
                    false, aHi[mi].bf, false, bLo[ni].bf,
                    (short)0, acc[mi][ni], false, false);
                acc[mi][ni] = __builtin_amdgcn_wmma_f32_16x16x32_bf16(
                    false, aLo[mi].bf, false, bHi[ni].bf,
                    (short)0, acc[mi][ni], false, false);
            }
        }
    }

    // ---- Epilogue: out[m,n] = acc * emb[m,n]
    // C layout (f32 16x16): lane -> col = lane&15; VGPR v -> row v + 8*(lane>=16)
#pragma unroll
    for (int mi = 0; mi < 2; ++mi) {
#pragma unroll
        for (int ni = 0; ni < 4; ++ni) {
            const int n     = n_blk + wn * 64 + ni * 16 + lane15;
            const int mbase = m_wave + mi * 16 + laneHi * 8;
#pragma unroll
            for (int v = 0; v < 8; ++v) {
                size_t off = (size_t)(mbase + v) * NDIM + n;
                out[off] = acc[mi][ni][v] * emb[off];
            }
        }
    }
}

// ---------------------------------------------------------------------------
// Fallback GEMM: converts A fp32->bf16 hi/lo inside the loop (small ws).
// ---------------------------------------------------------------------------
__global__ __launch_bounds__(256)
void bilinear_wmma_convert_kernel(const float* __restrict__ emb,
                                  const unsigned short* __restrict__ WtHi,
                                  const unsigned short* __restrict__ WtLo,
                                  float* __restrict__ out) {
    const int lane   = threadIdx.x & 31;
    const int wave   = threadIdx.x >> 5;
    const int wm     = wave >> 2;
    const int wn     = wave & 3;
    const int lane15 = lane & 15;
    const int laneHi = lane >> 4;

    const int m_wave = blockIdx.y * 64 + wm * 32;
    const int n_wave = blockIdx.x * 128 + wn * 32;

    const v8f vzero = {0.f, 0.f, 0.f, 0.f, 0.f, 0.f, 0.f, 0.f};
    v8f acc[2][2];
    acc[0][0] = vzero; acc[0][1] = vzero;
    acc[1][0] = vzero; acc[1][1] = vzero;

    const int a_kc = laneHi * 8;
    const int b_kc = laneHi * 8;

    for (int k = 0; k < KDIM; k += 32) {
        Frag16 aHi[2], aLo[2], bHi[2], bLo[2];
#pragma unroll
        for (int mi = 0; mi < 2; ++mi) {
            const float* p =
                emb + (size_t)(m_wave + mi * 16 + lane15) * KDIM + k + a_kc;
            float4 x0 = *(const float4*)(p);
            float4 x1 = *(const float4*)(p + 4);
            float4 x2 = *(const float4*)(p + 16);
            float4 x3 = *(const float4*)(p + 20);
            float fv[16] = {x0.x, x0.y, x0.z, x0.w, x1.x, x1.y, x1.z, x1.w,
                            x2.x, x2.y, x2.z, x2.w, x3.x, x3.y, x3.z, x3.w};
#pragma unroll
            for (int i = 0; i < 16; ++i) {
                unsigned short h = f32_to_bf16_rn(fv[i]);
                aHi[mi].u[i] = h;
                aLo[mi].u[i] = f32_to_bf16_rn(fv[i] - bf16_to_f32(h));
            }
        }
#pragma unroll
        for (int ni = 0; ni < 2; ++ni) {
            size_t roff = (size_t)(n_wave + ni * 16 + lane15) * KDIM + k + b_kc;
            const unsigned short* ph = WtHi + roff;
            const unsigned short* pl = WtLo + roff;
            bHi[ni].q[0] = *(const uint4*)(ph);
            bHi[ni].q[1] = *(const uint4*)(ph + 16);
            bLo[ni].q[0] = *(const uint4*)(pl);
            bLo[ni].q[1] = *(const uint4*)(pl + 16);
        }
#pragma unroll
        for (int mi = 0; mi < 2; ++mi) {
#pragma unroll
            for (int ni = 0; ni < 2; ++ni) {
                acc[mi][ni] = __builtin_amdgcn_wmma_f32_16x16x32_bf16(
                    false, aHi[mi].bf, false, bHi[ni].bf,
                    (short)0, acc[mi][ni], false, false);
                acc[mi][ni] = __builtin_amdgcn_wmma_f32_16x16x32_bf16(
                    false, aHi[mi].bf, false, bLo[ni].bf,
                    (short)0, acc[mi][ni], false, false);
                acc[mi][ni] = __builtin_amdgcn_wmma_f32_16x16x32_bf16(
                    false, aLo[mi].bf, false, bHi[ni].bf,
                    (short)0, acc[mi][ni], false, false);
            }
        }
    }

#pragma unroll
    for (int mi = 0; mi < 2; ++mi) {
#pragma unroll
        for (int ni = 0; ni < 2; ++ni) {
            const int n     = n_wave + ni * 16 + lane15;
            const int mbase = m_wave + mi * 16 + laneHi * 8;
#pragma unroll
            for (int v = 0; v < 8; ++v) {
                size_t off = (size_t)(mbase + v) * NDIM + n;
                out[off] = acc[mi][ni][v] * emb[off];
            }
        }
    }
}

// ---------------------------------------------------------------------------
extern "C" void kernel_launch(void* const* d_in, const int* in_sizes, int n_in,
                              void* d_out, int out_size, void* d_ws, size_t ws_size,
                              hipStream_t stream) {
    (void)in_sizes; (void)n_in; (void)out_size;

    const float* emb = (const float*)d_in[0];          // [16384,40,32] fp32
    const float* W   = (const float*)d_in[1];          // [40,40,32,32] fp32
    float*       out = (float*)d_out;                  // [16384,40,32] fp32

    const size_t wElems = (size_t)KDIM * NDIM;
    const size_t aElems = (size_t)BATCH * KDIM;
    const size_t wBytes = 2 * wElems * sizeof(unsigned short);   // ~6.55 MB
    const size_t aBytes = 2 * aElems * sizeof(unsigned short);   // ~83.9 MB

    unsigned short* WtHi = (unsigned short*)d_ws;
    unsigned short* WtLo = WtHi + wElems;

    prep_w_kernel<<<(int)((wElems + 255) / 256), 256, 0, stream>>>(W, WtHi, WtLo);

    if (ws_size >= wBytes + aBytes) {
        unsigned short* AHi = WtLo + wElems;
        unsigned short* ALo = AHi + aElems;
        prep_emb_kernel<<<(int)((aElems + 255) / 256), 256, 0, stream>>>(emb, AHi, ALo);

        dim3 grid(NDIM / 256, BATCH / 64);   // (5, 256)
        bilinear_wmma_tdm_kernel<<<grid, 256, 0, stream>>>(
            emb, AHi, ALo, WtHi, WtLo, out);
    } else {
        dim3 grid(NDIM / 128, BATCH / 64);   // (10, 256)
        bilinear_wmma_convert_kernel<<<grid, 256, 0, stream>>>(
            emb, WtHi, WtLo, out);
    }
}
